// GroupedQueryAttention_33767032881425
// MI455X (gfx1250) — compile-verified, matrix operand order checked by reading, not run on hardware
//
#include <hip/hip_runtime.h>
#include <hip/hip_bf16.h>

#define HIDDEN 4096
#define SEQ    2048
#define BATCH  2
#define NHQ    32
#define NHKV   8
#define HD     128

typedef __attribute__((ext_vector_type(16))) __bf16 v16bf;
typedef __attribute__((ext_vector_type(8)))  __bf16 v8bf;
typedef __attribute__((ext_vector_type(8)))  float  v8f;
typedef __attribute__((ext_vector_type(4)))  unsigned int v4u;
typedef __attribute__((ext_vector_type(8)))  int v8i;
typedef __attribute__((ext_vector_type(4)))  int v4i;

typedef __attribute__((address_space(1))) v4i as1_v4i;   // global
typedef __attribute__((address_space(3))) v4i as3_v4i;   // LDS

union Frag16 { v16bf v; v8bf h[2]; };

// CDNA5 async data-mover availability (device pass only; host pass parses fallback)
#if defined(__gfx1250__) && __has_builtin(__builtin_amdgcn_tensor_load_to_lds)
#define USE_TDM 1
#else
#define USE_TDM 0
#endif
#if defined(__gfx1250__) && __has_builtin(__builtin_amdgcn_global_load_async_to_lds_b128)
#define USE_ASYNC 1
#else
#define USE_ASYNC 0
#endif

__device__ __forceinline__ void wait_async0() {
#if defined(__gfx1250__) && __has_builtin(__builtin_amdgcn_s_wait_asynccnt)
    __builtin_amdgcn_s_wait_asynccnt(0);
#elif defined(__gfx1250__)
    asm volatile("s_wait_asynccnt 0" ::: "memory");
#endif
}
__device__ __forceinline__ void wait_tensor0() {
#if defined(__gfx1250__) && __has_builtin(__builtin_amdgcn_s_wait_tensorcnt)
    __builtin_amdgcn_s_wait_tensorcnt(0);
#elif defined(__gfx1250__)
    asm volatile("s_wait_tensorcnt 0" ::: "memory");
#endif
}

// ---------------------------------------------------------------------------
// GEMM: C[M,N] = A[M,K] * B[K,N], A fp32 or bf16, B fp32, C fp32.
// Block 256 thr (8 waves), tile 128x128, BK=32 (one v_wmma_f32_16x16x32_bf16
// K-step). LDS holds A as [m][k] and B transposed to [n][k] so both fragment
// loads are two contiguous ds_load_b128 per lane (CDNA5 16-bit A/B layout).
// bf16-A path stages its tile with GLOBAL_LOAD_ASYNC_TO_LDS_B128 (ASYNCcnt).
// ---------------------------------------------------------------------------
template<bool ABF16>
__global__ __launch_bounds__(256)
void gemm_bf16_wmma(const void* __restrict__ Araw, const float* __restrict__ B,
                    float* __restrict__ C, int M, int N, int K)
{
    constexpr int BM = 128, BN = 128, BK = 32, LDT = 56; // 112B row stride: 16B aligned, conflict-free
    __shared__ __bf16 As[BM][LDT];
    __shared__ __bf16 Bs[BN][LDT];

    const int tid   = threadIdx.x;
    const int lane  = tid & 31;
    const int wave  = tid >> 5;
    const int lrow  = lane & 15;
    const int lhalf = lane >> 4;
    const int wm    = (wave & 1) * 64;   // 2 waves along M -> 64 rows each
    const int wn    = (wave >> 1) * 32;  // 4 waves along N -> 32 cols each
    const int m0    = blockIdx.y * BM;
    const int n0    = blockIdx.x * BN;

    v8f acc[4][2];
#pragma unroll
    for (int i = 0; i < 4; ++i)
#pragma unroll
        for (int j = 0; j < 2; ++j)
            acc[i][j] = (v8f)0.0f;

    for (int k0 = 0; k0 < K; k0 += BK) {
        // ---- stage A tile ----
        if constexpr (ABF16) {
            const __bf16* A = (const __bf16*)Araw;
#pragma unroll
            for (int i = 0; i < 2; ++i) {
                int idx = tid + i * 256;          // 512 slots of 8 bf16
                int r = idx >> 2, cg = (idx & 3) * 8;
                const __bf16* gp = A + (size_t)(m0 + r) * K + k0 + cg;
#if USE_ASYNC
                __builtin_amdgcn_global_load_async_to_lds_b128(
                    (as1_v4i*)(unsigned long long)gp,
                    (as3_v4i*)(unsigned long long)&As[r][cg], 0, 0);
#else
                *(uint4*)&As[r][cg] = *(const uint4*)gp;
#endif
            }
        } else {
            const float* A = (const float*)Araw;
#pragma unroll
            for (int i = 0; i < 4; ++i) {
                int idx = tid + i * 256;          // 1024 slots of float4
                int r = idx >> 3, cg = (idx & 7) * 4;
                float4 f = *(const float4*)(A + (size_t)(m0 + r) * K + k0 + cg);
                As[r][cg + 0] = (__bf16)f.x;
                As[r][cg + 1] = (__bf16)f.y;
                As[r][cg + 2] = (__bf16)f.z;
                As[r][cg + 3] = (__bf16)f.w;
            }
        }
        // ---- stage B tile transposed: Bs[n][k] (fp32 -> bf16 convert) ----
#pragma unroll
        for (int i = 0; i < 4; ++i) {
            int idx = tid + i * 256;              // 32 k-rows x 32 float4 groups
            int kr = idx >> 5;
            int ng = (idx & 31) * 4;
            float4 f = *(const float4*)(B + (size_t)(k0 + kr) * N + n0 + ng);
            Bs[ng + 0][kr] = (__bf16)f.x;
            Bs[ng + 1][kr] = (__bf16)f.y;
            Bs[ng + 2][kr] = (__bf16)f.z;
            Bs[ng + 3][kr] = (__bf16)f.w;
        }
        if (k0 + BK < K) {
            // hint next A tile toward the caches (global_prefetch_b8)
            const char* pa = (const char*)Araw +
                ((size_t)(m0 + (tid >> 1)) * K + k0 + BK) * (ABF16 ? 2 : 4);
            __builtin_prefetch(pa, 0, 1);
        }
        if constexpr (ABF16) wait_async0();
        __syncthreads();

        // ---- fragments + 8 WMMAs per wave ----
        Frag16 af[4], bf[2];
#pragma unroll
        for (int i = 0; i < 4; ++i) {
            int r = wm + i * 16 + lrow;
            af[i].h[0] = *(const v8bf*)&As[r][lhalf * 8];
            af[i].h[1] = *(const v8bf*)&As[r][16 + lhalf * 8];
        }
#pragma unroll
        for (int j = 0; j < 2; ++j) {
            int c = wn + j * 16 + lrow;
            bf[j].h[0] = *(const v8bf*)&Bs[c][lhalf * 8];
            bf[j].h[1] = *(const v8bf*)&Bs[c][16 + lhalf * 8];
        }
#pragma unroll
        for (int i = 0; i < 4; ++i)
#pragma unroll
            for (int j = 0; j < 2; ++j)
                acc[i][j] = __builtin_amdgcn_wmma_f32_16x16x32_bf16(
                    false, af[i].v, false, bf[j].v, (short)0, acc[i][j], false, false);
        __syncthreads();
    }

    // ---- epilogue: C/D layout -> element e is row e (+8 for hi lane-half) ----
#pragma unroll
    for (int i = 0; i < 4; ++i)
#pragma unroll
        for (int j = 0; j < 2; ++j)
#pragma unroll
            for (int e = 0; e < 8; ++e) {
                int m = m0 + wm + i * 16 + e + 8 * lhalf;
                int n = n0 + wn + j * 16 + lrow;
                C[(size_t)m * N + n] = acc[i][j][e];
            }
}

// ---------------------------------------------------------------------------
// RoPE (+ cast) with layout change [b][s][h][d] fp32 -> [b][h][s][d] bf16
// ---------------------------------------------------------------------------
__global__ __launch_bounds__(256)
void rope_cast(const float* __restrict__ src, __bf16* __restrict__ dst,
               int nheads, int rowStride, int doRope)
{
    int idx = blockIdx.x * blockDim.x + threadIdx.x;  // over b*nheads*SEQ*64 pairs
    int i = idx & 63;
    int s = (idx >> 6) & (SEQ - 1);
    int h = (idx >> 17) % nheads;
    int b = (idx >> 17) / nheads;
    size_t so = (size_t)(b * SEQ + s) * rowStride + h * HD + i;
    float x1 = src[so], x2 = src[so + 64];
    float o1 = x1, o2 = x2;
    if (doRope) {
        float inv = __powf(10000.0f, -(float)i * (1.0f / 64.0f));
        float fr  = (float)s * inv;
        float sn, c;
        __sincosf(fr, &sn, &c);
        o1 = x1 * c - x2 * sn;
        o2 = x2 * c + x1 * sn;
    }
    size_t doff = ((size_t)(b * nheads + h) * SEQ + s) * HD + i;
    dst[doff]      = (__bf16)o1;
    dst[doff + 64] = (__bf16)o2;
}

// ---------------------------------------------------------------------------
// Flash-style causal GQA attention. Grid: (BATCH*NHQ, SEQ/128).
// 8 waves x 16 query rows; key tiles of 64. Q frags straight from global
// (already K-contiguous); K stages via the Tensor Data Mover into LDS with
// hardware pad (row = 64 DW + 4 DW pad == 136-element stride); V transposed
// to [d][key]; P re-striped through LDS into A-frag layout.
// ---------------------------------------------------------------------------
__global__ __launch_bounds__(256)
void attn_kernel(const __bf16* __restrict__ Qr, const __bf16* __restrict__ Kr,
                 const __bf16* __restrict__ Vr, __bf16* __restrict__ Ob)
{
    constexpr int BQ = 128, BKV = 64, LDK = 136, LDV = 72, LDP = 72;
    __shared__ __bf16 Ks[BKV][LDK];  // [key][d]
    __shared__ __bf16 Vt[HD][LDV];   // [d][key]
    __shared__ __bf16 Ps[BQ][LDP];   // [q][key]

    const int tid   = threadIdx.x;
    const int lane  = tid & 31;
    const int wave  = tid >> 5;
    const int lrow  = lane & 15;
    const int lhalf = lane >> 4;
    const int bh = blockIdx.x;
    const int b  = bh >> 5;
    const int h  = bh & 31;
    const int kh = h >> 2;          // 4 query heads per KV head
    const int qt = blockIdx.y;

    const __bf16* Qbase = Qr + (size_t)bh * SEQ * HD;
    const __bf16* Kbase = Kr + (size_t)(b * NHKV + kh) * SEQ * HD;
    const __bf16* Vbase = Vr + (size_t)(b * NHKV + kh) * SEQ * HD;

    // preload Q fragments (16 rows x 128 d = 4 K-chunks)
    Frag16 qf[4];
    const int qrow = qt * BQ + wave * 16 + lrow;
#pragma unroll
    for (int kc = 0; kc < 4; ++kc) {
        const __bf16* p = Qbase + (size_t)qrow * HD + kc * 32 + lhalf * 8;
        qf[kc].h[0] = *(const v8bf*)p;
        qf[kc].h[1] = *(const v8bf*)(p + 16);
    }

    v8f oacc[8];
#pragma unroll
    for (int d = 0; d < 8; ++d) oacc[d] = (v8f)0.0f;
    float mrun[8], lrun[8];
#pragma unroll
    for (int e = 0; e < 8; ++e) { mrun[e] = -3.0e38f; lrun[e] = 0.0f; }

    const int q_e_base = qt * BQ + wave * 16 + 8 * lhalf; // q index of element e = base+e
    const float scale = 0.08838834764831845f;             // 1/sqrt(128)
    const int nkt = 2 * qt + 2;                           // causal key-tile count

    for (int kt = 0; kt < nkt; ++kt) {
        const int kbase = kt * BKV;

        // ---- stage K tile: TDM descriptor load (TENSORcnt) or sync fallback ----
#if USE_TDM
        if (wave == 0) {
            unsigned long long ga =
                (unsigned long long)(const void*)(Kbase + (size_t)kbase * HD);
            unsigned ldsa = (unsigned)(unsigned long long)(const void*)&Ks[0][0];
            v4u g0;
            g0.x = 1u;                                   // count=1, user desc
            g0.y = ldsa;                                 // lds_addr (bytes)
            g0.z = (unsigned)(ga & 0xffffffffu);         // global_addr[31:0]
            g0.w = (unsigned)((ga >> 32) & 0x01ffffffu)  // global_addr[56:32]
                   | (2u << 30);                         // type=2 (image)
            v8i g1;
            g1[0] = (int)((1u << 16)     // data_size = 2 bytes
                        | (1u << 20)     // pad_enable
                        | (5u << 22)     // pad_interval: every 64 DWORDs (one 128-elem row)
                        | (3u << 25));   // pad_amount: 4 DWORDs (8 bf16) -> 136-elem stride
            g1[1] = (int)(128u << 16);   // tensor_dim0 = HD (low16 @ bits 63:48->[31:16])
            g1[2] = (int)((unsigned)SEQ << 16); // tensor_dim1 = SEQ (low16 @ [95:80])
            g1[3] = (int)(128u << 16);   // tile_dim0 = HD  (bits 127:112)
            g1[4] = (int)BKV;            // tile_dim1 = 64 keys (bits 143:128)
            g1[5] = (int)HD;             // tensor_dim0_stride = HD (bits 191:160)
            g1[6] = 0;
            g1[7] = 0;
            v4i gz; gz[0] = gz[1] = gz[2] = gz[3] = 0;   // 2-D: groups 2/3 unused
#if __clang_major__ >= 23
            v8i gz8; gz8[0]=gz8[1]=gz8[2]=gz8[3]=gz8[4]=gz8[5]=gz8[6]=gz8[7]=0;
            __builtin_amdgcn_tensor_load_to_lds(g0, g1, gz, gz, gz8, 0);
#else
            __builtin_amdgcn_tensor_load_to_lds(g0, g1, gz, gz, 0);
#endif
        }
#else
#pragma unroll
        for (int i = 0; i < 4; ++i) {
            int idx = tid + i * 256;          // 64 rows x 16 groups of 8 bf16
            int r = idx >> 4, cg = (idx & 15) * 8;
            *(uint4*)&Ks[r][cg] =
                *(const uint4*)(Kbase + (size_t)(kbase + r) * HD + cg);
        }
#endif
        // ---- stage V tile transposed: Vt[d][key] ----
#pragma unroll
        for (int i = 0; i < 4; ++i) {
            int idx = tid + i * 256;
            int r = idx >> 4, cg = (idx & 15) * 8;
            v8bf v = *(const v8bf*)(Vbase + (size_t)(kbase + r) * HD + cg);
#pragma unroll
            for (int j = 0; j < 8; ++j) Vt[cg + j][r] = v[j];
        }
#if USE_TDM
        if (wave == 0) wait_tensor0();    // TDM done -> LDS valid for this wave
#endif
        __syncthreads();                  // publish Ks/Vt to all waves

        // S = Q K^T  (16 q x 64 k per wave)
        v8f sacc[4];
#pragma unroll
        for (int ns = 0; ns < 4; ++ns) sacc[ns] = (v8f)0.0f;
#pragma unroll
        for (int ns = 0; ns < 4; ++ns) {
            const int krow = ns * 16 + lrow;
#pragma unroll
            for (int kc = 0; kc < 4; ++kc) {
                Frag16 kf;
                kf.h[0] = *(const v8bf*)&Ks[krow][kc * 32 + lhalf * 8];
                kf.h[1] = *(const v8bf*)&Ks[krow][kc * 32 + 16 + lhalf * 8];
                sacc[ns] = __builtin_amdgcn_wmma_f32_16x16x32_bf16(
                    false, qf[kc].v, false, kf.v, (short)0, sacc[ns], false, false);
            }
        }

        // scale + causal mask + row max (16-lane groups align with C-layout rows)
        float rmax[8];
#pragma unroll
        for (int e = 0; e < 8; ++e) rmax[e] = -3.0e38f;
#pragma unroll
        for (int ns = 0; ns < 4; ++ns)
#pragma unroll
            for (int e = 0; e < 8; ++e) {
                float s = sacc[ns][e] * scale;
                int kg = kbase + ns * 16 + lrow;
                if (kg > q_e_base + e) s = -__builtin_inff();
                sacc[ns][e] = s;
                rmax[e] = fmaxf(rmax[e], s);
            }
#pragma unroll
        for (int e = 0; e < 8; ++e) {
            rmax[e] = fmaxf(rmax[e], __shfl_xor(rmax[e], 1, 32));
            rmax[e] = fmaxf(rmax[e], __shfl_xor(rmax[e], 2, 32));
            rmax[e] = fmaxf(rmax[e], __shfl_xor(rmax[e], 4, 32));
            rmax[e] = fmaxf(rmax[e], __shfl_xor(rmax[e], 8, 32));
        }
        float corr[8], rsum[8];
#pragma unroll
        for (int e = 0; e < 8; ++e) {
            float mnew = fmaxf(mrun[e], rmax[e]);
            corr[e] = __expf(mrun[e] - mnew);
            mrun[e] = mnew;
            rsum[e] = 0.0f;
        }
        // P = exp(S - m), spill to LDS for A-fragment re-striping
#pragma unroll
        for (int ns = 0; ns < 4; ++ns)
#pragma unroll
            for (int e = 0; e < 8; ++e) {
                float p = __expf(sacc[ns][e] - mrun[e]);
                rsum[e] += p;
                Ps[wave * 16 + e + 8 * lhalf][ns * 16 + lrow] = (__bf16)p;
            }
#pragma unroll
        for (int e = 0; e < 8; ++e) {
            rsum[e] += __shfl_xor(rsum[e], 1, 32);
            rsum[e] += __shfl_xor(rsum[e], 2, 32);
            rsum[e] += __shfl_xor(rsum[e], 4, 32);
            rsum[e] += __shfl_xor(rsum[e], 8, 32);
            lrun[e] = lrun[e] * corr[e] + rsum[e];
        }
#pragma unroll
        for (int d = 0; d < 8; ++d)
#pragma unroll
            for (int e = 0; e < 8; ++e) oacc[d][e] *= corr[e];

        // O += P V (per-wave LDS RAW: DS ops are in-order within a wave)
        const int prow = wave * 16 + lrow;
#pragma unroll
        for (int ds = 0; ds < 8; ++ds) {
            const int drow = ds * 16 + lrow;
#pragma unroll
            for (int kc = 0; kc < 2; ++kc) {
                Frag16 pf, vf;
                pf.h[0] = *(const v8bf*)&Ps[prow][kc * 32 + lhalf * 8];
                pf.h[1] = *(const v8bf*)&Ps[prow][kc * 32 + 16 + lhalf * 8];
                vf.h[0] = *(const v8bf*)&Vt[drow][kc * 32 + lhalf * 8];
                vf.h[1] = *(const v8bf*)&Vt[drow][kc * 32 + 16 + lhalf * 8];
                oacc[ds] = __builtin_amdgcn_wmma_f32_16x16x32_bf16(
                    false, pf.v, false, vf.v, (short)0, oacc[ds], false, false);
            }
        }
        __syncthreads();
    }

    // epilogue: O / l, write bf16 into [b*s][HIDDEN] at column h*128+d
#pragma unroll
    for (int ds = 0; ds < 8; ++ds)
#pragma unroll
        for (int e = 0; e < 8; ++e) {
            int row = qt * BQ + wave * 16 + e + 8 * lhalf;
            int col = h * HD + ds * 16 + lrow;
            Ob[(size_t)(b * SEQ + row) * HIDDEN + col] =
                (__bf16)(oacc[ds][e] / lrun[e]);
        }
}

// ---------------------------------------------------------------------------
extern "C" void kernel_launch(void* const* d_in, const int* in_sizes, int n_in,
                              void* d_out, int out_size, void* d_ws, size_t ws_size,
                              hipStream_t stream) {
    const float* x  = (const float*)d_in[0];
    const float* Wq = (const float*)d_in[1];
    const float* Wk = (const float*)d_in[2];
    const float* Wv = (const float*)d_in[3];
    const float* Wo = (const float*)d_in[4];
    float* out = (float*)d_out;

    char* ws = (char*)d_ws;
    float*  Qf = (float*) (ws);                         // 4096x4096 f32 = 64MB
    float*  Kf = (float*) (ws + ((size_t)64  << 20));   // 4096x1024 f32 = 16MB
    float*  Vf = (float*) (ws + ((size_t)80  << 20));   // 16MB
    __bf16* Qr = (__bf16*)(ws + ((size_t)96  << 20));   // [2][32][2048][128] = 32MB
    __bf16* Kr = (__bf16*)(ws + ((size_t)128 << 20));   // [2][8][2048][128]  = 8MB
    __bf16* Vr = (__bf16*)(ws + ((size_t)136 << 20));   // 8MB
    __bf16* Ob = (__bf16*)(ws + ((size_t)144 << 20));   // 4096x4096 bf16 = 32MB

    dim3 blk(256);
    const int M = BATCH * SEQ;  // 4096

    gemm_bf16_wmma<false><<<dim3(HIDDEN / 128, M / 128), blk, 0, stream>>>(x, Wq, Qf, M, HIDDEN, HIDDEN);
    gemm_bf16_wmma<false><<<dim3((NHKV * HD) / 128, M / 128), blk, 0, stream>>>(x, Wk, Kf, M, NHKV * HD, HIDDEN);
    gemm_bf16_wmma<false><<<dim3((NHKV * HD) / 128, M / 128), blk, 0, stream>>>(x, Wv, Vf, M, NHKV * HD, HIDDEN);

    int tq = BATCH * NHQ  * SEQ * 64;
    int tk = BATCH * NHKV * SEQ * 64;
    rope_cast<<<tq / 256, 256, 0, stream>>>(Qf, Qr, NHQ,  HIDDEN,    1);
    rope_cast<<<tk / 256, 256, 0, stream>>>(Kf, Kr, NHKV, NHKV * HD, 1);
    rope_cast<<<tk / 256, 256, 0, stream>>>(Vf, Vr, NHKV, NHKV * HD, 0);

    attn_kernel<<<dim3(BATCH * NHQ, SEQ / 128), blk, 0, stream>>>(Qr, Kr, Vr, Ob);

    gemm_bf16_wmma<true><<<dim3(HIDDEN / 128, M / 128), blk, 0, stream>>>(Ob, Wo, out, M, HIDDEN, HIDDEN);
}
